// GNN_16999480557858
// MI455X (gfx1250) — matrix operand order repaired and checked by reading, not compiled, not run on hardware
//
#include <hip/hip_runtime.h>

#define N_NODES  50000
#define N_EDGES  800000
#define DIM      128
#define N_GRAPHS 512
#define MTILE    5                     // 16-row subtiles per block: 80 rows/block
#define ROWS_PER_BLOCK (MTILE * 16)    // 50000 = 625 * 80

typedef __attribute__((ext_vector_type(16))) __bf16 v16bf;
typedef __attribute__((ext_vector_type(8)))  __bf16 v8bf;
typedef __attribute__((ext_vector_type(8)))  float  v8f;

// ---------------- init: deg=1 (self loop), pooled=0, cnt=0 ----------------
__global__ void gnn_init_kernel(float* __restrict__ deg, float* __restrict__ pooled,
                                float* __restrict__ cnt) {
  int i = blockIdx.x * blockDim.x + threadIdx.x;
  if (i < N_NODES)        deg[i] = 1.0f;
  if (i < N_GRAPHS * DIM) pooled[i] = 0.0f;
  if (i < N_GRAPHS)       cnt[i] = 0.0f;
}

// ---------------- degree accumulation over edges ----------------
__global__ void gnn_deg_kernel(const int* __restrict__ dst, float* __restrict__ deg) {
  int e = blockIdx.x * blockDim.x + threadIdx.x;
  if (e < N_EDGES) atomicAdd(&deg[dst[e]], 1.0f);
}

__global__ void gnn_dinv_kernel(const float* __restrict__ deg, float* __restrict__ dinv) {
  int i = blockIdx.x * blockDim.x + threadIdx.x;
  if (i < N_NODES) dinv[i] = rsqrtf(deg[i]);
}

// ---------------- fp32 -> bf16 conversion (for GEMM A operand) ----------------
__global__ void gnn_cvt_bf16_kernel(const float* __restrict__ in,
                                    __bf16* __restrict__ out) {
  int i = (blockIdx.x * blockDim.x + threadIdx.x) * 4;
  if (i >= N_NODES * DIM) return;
  const float4 v = *reinterpret_cast<const float4*>(in + i);
  out[i + 0] = (__bf16)v.x;
  out[i + 1] = (__bf16)v.y;
  out[i + 2] = (__bf16)v.z;
  out[i + 3] = (__bf16)v.w;
}

// ---------------- fused GEMM + dinv scale: hs = dinv[n] * (X @ W) ----------------
// Block = 256 threads (8 waves) computes 80 rows x 128 cols. X is bf16.
// A fragment = two aligned 16B bf16 loads (K {0..15} / +16 per half-wave).
// B fragment = one aligned 32B LDS load from pre-swizzled W (2x ds_load_b128).
__global__ __launch_bounds__(256) void gnn_gemm_scale_kernel(
    const __bf16* __restrict__ X, const float* __restrict__ W,
    const float* __restrict__ dinv,
    float* __restrict__ hs, float* __restrict__ acc) {
  __shared__ __attribute__((aligned(32))) __bf16 sWs[DIM * DIM];  // 32 KB

  // cooperative fill: coalesced global reads, swizzled LDS writes
  #pragma unroll
  for (int t = 0; t < (DIM * DIM) / 256; ++t) {
    int idx = t * 256 + threadIdx.x;
    int k  = idx >> 7;         // W row (K index)
    int cc = idx & 127;        // W col
    int slot = ((k >> 4) * DIM + cc) * 16 + (k & 15);
    sWs[slot] = (__bf16)W[idx];
  }
  __syncthreads();

  const int wave = threadIdx.x >> 5;    // column tile 0..7
  const int lane = threadIdx.x & 31;
  const int half = lane >> 4;
  const int lrow = lane & 15;
  const int coln = wave * 16 + lrow;    // output / B column this lane owns
  const int mbase = blockIdx.x * ROWS_PER_BLOCK;

  v8f c[MTILE];
  #pragma unroll
  for (int mt = 0; mt < MTILE; ++mt) c[mt] = (v8f){};

  #pragma unroll
  for (int kb = 0; kb < 4; ++kb) {
    const int kbase = kb * 32;
    // B fragment: one 32B LDS load per half-wave group
    const v16bf b = *reinterpret_cast<const v16bf*>(
        &sWs[((kb * 2 + half) * DIM + coln) * 16]);
    #pragma unroll
    for (int mt = 0; mt < MTILE; ++mt) {
      const __bf16* ap = X + (size_t)(mbase + mt * 16 + lrow) * DIM + kbase + half * 16;
      const v8bf lo = *reinterpret_cast<const v8bf*>(ap);       // K = kbase+16h+0..7
      const v8bf hi = *reinterpret_cast<const v8bf*>(ap + 8);   // K = kbase+16h+8..15
      const v16bf a = __builtin_shufflevector(lo, hi,
          0, 1, 2, 3, 4, 5, 6, 7, 8, 9, 10, 11, 12, 13, 14, 15);
      c[mt] = __builtin_amdgcn_wmma_f32_16x16x32_bf16(false, a, false, b,
                                                      (short)0, c[mt], false, false);
    }
  }

  // C/D layout: VGPR g -> M = g + 8*half, lane%16 -> N
  #pragma unroll
  for (int mt = 0; mt < MTILE; ++mt) {
    #pragma unroll
    for (int g = 0; g < 8; ++g) {
      int r = mbase + mt * 16 + half * 8 + g;
      float v = c[mt][g] * dinv[r];
      size_t o = (size_t)r * DIM + coln;
      hs[o]  = v;   // scaled features for gather
      acc[o] = v;   // accumulator init == self-loop contribution
    }
  }
}

// ---------------- edge scatter: acc[dst] += hs[src] ----------------
__global__ void gnn_scatter_kernel(const float* __restrict__ hs,
                                   const int* __restrict__ src,
                                   const int* __restrict__ dst,
                                   float* __restrict__ acc) {
  int tid = blockIdx.x * blockDim.x + threadIdx.x;
  int e = tid >> 5;
  int c = tid & 31;
  if (e >= N_EDGES) return;
  int s = src[e], d = dst[e];
  const float4 v = *reinterpret_cast<const float4*>(hs + (size_t)s * DIM + c * 4);
  float* pa = acc + (size_t)d * DIM + c * 4;
  atomicAdd(pa + 0, v.x);
  atomicAdd(pa + 1, v.y);
  atomicAdd(pa + 2, v.z);
  atomicAdd(pa + 3, v.w);
}

// ---------------- out = relu(dinv[n]*acc + bias), fp32 + bf16 copies ----------------
__global__ void gnn_finalize_kernel(const float* __restrict__ acc,
                                    const float* __restrict__ dinv,
                                    const float* __restrict__ bias,
                                    float* __restrict__ out,
                                    __bf16* __restrict__ outbf) {
  int tid = blockIdx.x * blockDim.x + threadIdx.x;
  int n = tid >> 5;
  int c = tid & 31;
  if (n >= N_NODES) return;
  float dv = dinv[n];
  size_t o = (size_t)n * DIM + c * 4;
  const float4 a = *reinterpret_cast<const float4*>(acc + o);
  const float4 b = *reinterpret_cast<const float4*>(bias + c * 4);
  float4 r;
  r.x = fmaxf(fmaf(a.x, dv, b.x), 0.0f);
  r.y = fmaxf(fmaf(a.y, dv, b.y), 0.0f);
  r.z = fmaxf(fmaf(a.z, dv, b.z), 0.0f);
  r.w = fmaxf(fmaf(a.w, dv, b.w), 0.0f);
  *reinterpret_cast<float4*>(out + o) = r;
  outbf[o + 0] = (__bf16)r.x;
  outbf[o + 1] = (__bf16)r.y;
  outbf[o + 2] = (__bf16)r.z;
  outbf[o + 3] = (__bf16)r.w;
}

// ---------------- segment-sum pooling ----------------
__global__ void gnn_pool_kernel(const float* __restrict__ h,
                                const int* __restrict__ batch,
                                float* __restrict__ pooled,
                                float* __restrict__ cnt) {
  int tid = blockIdx.x * blockDim.x + threadIdx.x;
  int n = tid >> 5;
  int c = tid & 31;
  if (n >= N_NODES) return;
  int g = batch[n];
  const float4 v = *reinterpret_cast<const float4*>(h + (size_t)n * DIM + c * 4);
  float* pp = pooled + (size_t)g * DIM + c * 4;
  atomicAdd(pp + 0, v.x);
  atomicAdd(pp + 1, v.y);
  atomicAdd(pp + 2, v.z);
  atomicAdd(pp + 3, v.w);
  if (c == 0) atomicAdd(&cnt[g], 1.0f);
}

// ---------------- readout: out[g] = dot(pooled[g],Wl)/max(cnt,1) + bl ----------------
__global__ void gnn_readout_kernel(const float* __restrict__ pooled,
                                   const float* __restrict__ cnt,
                                   const float* __restrict__ Wl,
                                   const float* __restrict__ bl,
                                   float* __restrict__ out) {
  int g = blockIdx.x * 4 + (threadIdx.x >> 5);
  int lane = threadIdx.x & 31;
  const float* pg = pooled + (size_t)g * DIM;
  float partial = 0.0f;
  #pragma unroll
  for (int j = 0; j < 4; ++j)
    partial = fmaf(pg[lane * 4 + j], Wl[lane * 4 + j], partial);
  #pragma unroll
  for (int off = 16; off > 0; off >>= 1)
    partial += __shfl_down(partial, off);   // wave32
  if (lane == 0) out[g] = partial / fmaxf(cnt[g], 1.0f) + bl[0];
}

extern "C" void kernel_launch(void* const* d_in, const int* in_sizes, int n_in,
                              void* d_out, int out_size, void* d_ws, size_t ws_size,
                              hipStream_t stream) {
  (void)in_sizes; (void)n_in; (void)out_size; (void)ws_size;
  const float* x    = (const float*)d_in[0];
  const int*   eidx = (const int*)d_in[1];
  const int*   src  = eidx;             // edge_index row 0
  const int*   dst  = eidx + N_EDGES;   // edge_index row 1
  const int*   batch = (const int*)d_in[2];
  const float* W1 = (const float*)d_in[3];
  const float* b1 = (const float*)d_in[4];
  const float* W2 = (const float*)d_in[5];
  const float* b2 = (const float*)d_in[6];
  const float* Wl = (const float*)d_in[7];
  const float* bl = (const float*)d_in[8];
  float* out = (float*)d_out;

  float* ws     = (float*)d_ws;
  float* deg    = ws;                                   // 50000
  float* dinv   = deg + N_NODES;                        // 50000
  float* bufA   = dinv + N_NODES;                       // 50000*128 fp32
  float* bufB   = bufA + (size_t)N_NODES * DIM;         // 50000*128 fp32
  float* bufC   = bufB + (size_t)N_NODES * DIM;         // 50000*128 fp32
  float* pooled = bufC + (size_t)N_NODES * DIM;         // 512*128
  float* cnt    = pooled + (size_t)N_GRAPHS * DIM;      // 512

  // bf16 activations aliased into bufC's region (fp32 bufC only live from finalize2 on)
  __bf16* xbf  = (__bf16*)bufC;                          // lives: cvt .. GEMM1
  __bf16* h1bf = (__bf16*)(bufC + (size_t)N_NODES * DIM / 2); // lives: fin1 .. GEMM2
  __bf16* dumbf = (__bf16*)bufA;                         // finalize2 bf16 sink (hs2 dead)

  // normalization
  gnn_init_kernel<<<(N_GRAPHS * DIM + 255) / 256, 256, 0, stream>>>(deg, pooled, cnt);
  gnn_deg_kernel<<<(N_EDGES + 255) / 256, 256, 0, stream>>>(dst, deg);
  gnn_dinv_kernel<<<(N_NODES + 255) / 256, 256, 0, stream>>>(deg, dinv);

  // x -> bf16
  gnn_cvt_bf16_kernel<<<(N_NODES * DIM / 4 + 255) / 256, 256, 0, stream>>>(x, xbf);

  // layer 1: hs1=bufA, acc1=bufB -> h1 (bf16) = h1bf
  gnn_gemm_scale_kernel<<<N_NODES / ROWS_PER_BLOCK, 256, 0, stream>>>(xbf, W1, dinv, bufA, bufB);
  gnn_scatter_kernel<<<(N_EDGES * 32) / 256, 256, 0, stream>>>(bufA, src, dst, bufB);
  gnn_finalize_kernel<<<(N_NODES * 32) / 256, 256, 0, stream>>>(bufB, dinv, b1, bufA, h1bf);

  // layer 2: hs2=bufA (over dead fp32 h1), acc2=bufB -> h2 fp32 = bufC
  gnn_gemm_scale_kernel<<<N_NODES / ROWS_PER_BLOCK, 256, 0, stream>>>(h1bf, W2, dinv, bufA, bufB);
  gnn_scatter_kernel<<<(N_EDGES * 32) / 256, 256, 0, stream>>>(bufA, src, dst, bufB);
  gnn_finalize_kernel<<<(N_NODES * 32) / 256, 256, 0, stream>>>(bufB, dinv, b2, bufC, dumbf);

  // pooling + readout
  gnn_pool_kernel<<<(N_NODES * 32) / 256, 256, 0, stream>>>(bufC, batch, pooled, cnt);
  gnn_readout_kernel<<<N_GRAPHS / 4, 128, 0, stream>>>(pooled, cnt, Wl, bl, out);
}
